// Attention_8555574854319
// MI455X (gfx1250) — matrix operand order repaired and checked by reading, not compiled, not run on hardware
//
#include <hip/hip_runtime.h>

typedef __attribute__((ext_vector_type(16))) __bf16 v16bf;
typedef __attribute__((ext_vector_type(8)))  __bf16 v8bf;
typedef __attribute__((ext_vector_type(8)))  float  v8f;

#define NH   12
#define TT   1024
#define CC   768
#define DD   64
#define NTOK (12 * 1024)   // B*T = 12288

__device__ __forceinline__ v8f wmma_bf16(v16bf a, v16bf b, v8f c) {
  // D = A(16x32 bf16) x B(32x16 bf16) + C(16x16 f32)
  return __builtin_amdgcn_wmma_f32_16x16x32_bf16(false, a, false, b, (short)0, c,
                                                 false, false);
}

// A-matrix (16x32, 16-bit) fragment: lane = M row (lm), half selects K 0-15/16-31
// contiguous (k = 16*half + e), matching the B fragments.
__device__ __forceinline__ v16bf load_a_frag(const __bf16* base, int ld, int row,
                                             int k0, int half) {
  const __bf16* p = base + (size_t)row * ld + k0 + half * 16;
  v8bf lo = *(const v8bf*)p;         // K = half*16 + 0..7
  v8bf hi = *(const v8bf*)(p + 8);   // K = half*16 + 8..15
  v16bf a;
#pragma unroll
  for (int i = 0; i < 8; ++i) { a[i] = lo[i]; a[8 + i] = hi[i]; }
  return a;
}

__global__ void f32_to_bf16_kernel(const float* __restrict__ s,
                                   __bf16* __restrict__ d, int n) {
  int i = blockIdx.x * blockDim.x + threadIdx.x;
  if (i < n) d[i] = (__bf16)s[i];
}

// C[M,N] = A[M,K](bf16) * W[N,K]^T(bf16) + bias[N]
// mode 0/1: store bf16 to (B,H,T,D); mode 2: store bf16 to (B,H,D,T);
// mode 3: store f32 row-major [M,N].
__global__ void gemm_bf16_kernel(const __bf16* __restrict__ A,
                                 const __bf16* __restrict__ W,
                                 const float* __restrict__ bias,
                                 void* __restrict__ out,
                                 int M, int N, int K, int mode) {
  int lane = threadIdx.x & 31;
  int wv   = threadIdx.x >> 5;
  int wid  = blockIdx.x * 4 + wv;
  int tilesN = N >> 5;
  int tm = wid / tilesN, tn = wid % tilesN;
  int m0 = tm * 32, n0 = tn * 32;
  if (m0 >= M) return;
  int lm = lane & 15, half = lane >> 4;

  v8f c00 = {}, c01 = {}, c10 = {}, c11 = {};
  for (int k0 = 0; k0 < K; k0 += 32) {
    v16bf a0 = load_a_frag(A, K, m0 + lm,      k0, half);
    v16bf a1 = load_a_frag(A, K, m0 + 16 + lm, k0, half);
    // B(k,n) = W[n][k]: lane = n, 16 contiguous k values per lane (k = e + 16*half)
    v16bf b0 = *(const v16bf*)(W + (size_t)(n0 + lm) * K      + k0 + half * 16);
    v16bf b1 = *(const v16bf*)(W + (size_t)(n0 + 16 + lm) * K + k0 + half * 16);
    c00 = wmma_bf16(a0, b0, c00);
    c01 = wmma_bf16(a0, b1, c01);
    c10 = wmma_bf16(a1, b0, c10);
    c11 = wmma_bf16(a1, b1, c11);
  }

  v8f   tiles[4] = {c00, c01, c10, c11};
  int   moff[4]  = {0, 0, 16, 16};
  int   noff[4]  = {0, 16, 0, 16};
#pragma unroll
  for (int t4 = 0; t4 < 4; ++t4) {
    v8f c = tiles[t4];
#pragma unroll
    for (int r = 0; r < 8; ++r) {
      int row = m0 + moff[t4] + r + 8 * half;  // token index n = b*T + t
      int col = n0 + noff[t4] + lm;            // channel = h*64 + d
      float v = c[r] + bias[col];
      if (mode == 3) {
        ((float*)out)[(size_t)row * N + col] = v;
      } else {
        int b = row >> 10, t = row & (TT - 1);
        int h = col >> 6,  d = col & (DD - 1);
        if (mode == 2)
          ((__bf16*)out)[((size_t)(b * NH + h) * DD + d) * TT + t] = (__bf16)v;
        else
          ((__bf16*)out)[((size_t)(b * NH + h) * TT + t) * DD + d] = (__bf16)v;
      }
    }
  }
}

// Flash attention: one wave per (b,h, 16-row q block); key blocks of 32.
// q,k: (B,H,T,D) bf16. vt: (B,H,D,T) bf16. y: (B,T,C) bf16.
__global__ void attn_kernel(const __bf16* __restrict__ q,
                            const __bf16* __restrict__ k,
                            const __bf16* __restrict__ vt,
                            __bf16* __restrict__ y) {
  __shared__ __bf16 Pl[4][16][32];  // per-wave P staging (C-layout -> A-layout)
  int lane = threadIdx.x & 31;
  int wv   = threadIdx.x >> 5;
  int wid  = blockIdx.x * 4 + wv;
  int qb   = wid & 63;          // 64 q-blocks of 16 rows
  int bh   = wid >> 6;          // 0..143
  int q0   = qb * 16;
  int lm = lane & 15, half = lane >> 4;

  const __bf16* qp = q  + (size_t)bh * TT * DD;
  const __bf16* kp = k  + (size_t)bh * TT * DD;
  const __bf16* vp = vt + (size_t)bh * DD * TT;

  v16bf aq0 = load_a_frag(qp, DD, q0 + lm, 0,  half);
  v16bf aq1 = load_a_frag(qp, DD, q0 + lm, 32, half);

  float m8[8], l8[8];
  v8f acc[4];
#pragma unroll
  for (int r = 0; r < 8; ++r) { m8[r] = -3.0e38f; l8[r] = 0.0f; }
#pragma unroll
  for (int dt = 0; dt < 4; ++dt) acc[dt] = (v8f){};

  const float scale = 0.125f;  // 1/sqrt(64)

  for (int j0 = 0; j0 < q0 + 16; j0 += 32) {
    v8f s0 = {}, s1 = {};
    {
      v16bf bk;
      bk = *(const v16bf*)(kp + (size_t)(j0 + lm) * DD      + half * 16);
      s0 = wmma_bf16(aq0, bk, s0);
      bk = *(const v16bf*)(kp + (size_t)(j0 + lm) * DD + 32 + half * 16);
      s0 = wmma_bf16(aq1, bk, s0);
      bk = *(const v16bf*)(kp + (size_t)(j0 + 16 + lm) * DD      + half * 16);
      s1 = wmma_bf16(aq0, bk, s1);
      bk = *(const v16bf*)(kp + (size_t)(j0 + 16 + lm) * DD + 32 + half * 16);
      s1 = wmma_bf16(aq1, bk, s1);
    }
    int key0 = j0 + lm, key1 = j0 + 16 + lm;
#pragma unroll
    for (int r = 0; r < 8; ++r) {
      int row = q0 + r + 8 * half;
      float v0 = (key0 <= row) ? s0[r] * scale : -3.0e38f;
      float v1 = (key1 <= row) ? s1[r] * scale : -3.0e38f;
      float mx = fmaxf(v0, v1);
      mx = fmaxf(mx, __shfl_xor(mx, 1, 32));
      mx = fmaxf(mx, __shfl_xor(mx, 2, 32));
      mx = fmaxf(mx, __shfl_xor(mx, 4, 32));
      mx = fmaxf(mx, __shfl_xor(mx, 8, 32));
      float mn = fmaxf(m8[r], mx);
      float al = __expf(m8[r] - mn);
      float p0 = __expf(v0 - mn);
      float p1 = __expf(v1 - mn);
      float rs = p0 + p1;
      rs += __shfl_xor(rs, 1, 32);
      rs += __shfl_xor(rs, 2, 32);
      rs += __shfl_xor(rs, 4, 32);
      rs += __shfl_xor(rs, 8, 32);
      l8[r] = l8[r] * al + rs;
      m8[r] = mn;
      acc[0][r] *= al; acc[1][r] *= al; acc[2][r] *= al; acc[3][r] *= al;
      Pl[wv][r + 8 * half][lm]      = (__bf16)p0;
      Pl[wv][r + 8 * half][16 + lm] = (__bf16)p1;
    }
    asm volatile("" ::: "memory");  // per-wave DS ops are in-order; block reordering
    v16bf ap;
#pragma unroll
    for (int e = 0; e < 16; ++e) {
      int kk = half * 16 + e;  // A-layout K index
      ap[e] = Pl[wv][lm][kk];
    }
    asm volatile("" ::: "memory");
#pragma unroll
    for (int dt = 0; dt < 4; ++dt) {
      // B(k=key, n=d): vt contiguous in key per fixed d
      v16bf bv = *(const v16bf*)(vp + (size_t)(dt * 16 + lm) * TT + j0 + half * 16);
      acc[dt] = wmma_bf16(ap, bv, acc[dt]);
    }
  }

  int b = bh / NH, h = bh % NH;
#pragma unroll
  for (int dt = 0; dt < 4; ++dt) {
#pragma unroll
    for (int r = 0; r < 8; ++r) {
      int row = q0 + r + 8 * half;
      float o = acc[dt][r] / l8[r];
      y[(size_t)(b * TT + row) * CC + h * DD + dt * 16 + lm] = (__bf16)o;
    }
  }
}

extern "C" void kernel_launch(void* const* d_in, const int* in_sizes, int n_in,
                              void* d_out, int out_size, void* d_ws, size_t ws_size,
                              hipStream_t stream) {
  const float* x  = (const float*)d_in[0];
  const float* Wq = (const float*)d_in[1];
  const float* bq = (const float*)d_in[2];
  const float* Wk = (const float*)d_in[3];
  const float* bk = (const float*)d_in[4];
  const float* Wv = (const float*)d_in[5];
  const float* bv = (const float*)d_in[6];
  const float* Wo = (const float*)d_in[7];
  const float* bo = (const float*)d_in[8];
  float* out = (float*)d_out;

  const size_t nX = (size_t)NTOK * CC;   // 9,437,184
  const size_t nW = (size_t)CC * CC;     //   589,824

  char* ws = (char*)d_ws;
  __bf16* xb  = (__bf16*)ws;  ws += nX * 2;
  __bf16* wqb = (__bf16*)ws;  ws += nW * 2;
  __bf16* wkb = (__bf16*)ws;  ws += nW * 2;
  __bf16* wvb = (__bf16*)ws;  ws += nW * 2;
  __bf16* wob = (__bf16*)ws;  ws += nW * 2;
  __bf16* qb  = (__bf16*)ws;  ws += nX * 2;  // (B,H,T,D)
  __bf16* kb  = (__bf16*)ws;  ws += nX * 2;  // (B,H,T,D)
  __bf16* vtb = (__bf16*)ws;  ws += nX * 2;  // (B,H,D,T)
  __bf16* yb  = (__bf16*)ws;  ws += nX * 2;  // (B,T,C)

  f32_to_bf16_kernel<<<(int)((nX + 255) / 256), 256, 0, stream>>>(x,  xb,  (int)nX);
  f32_to_bf16_kernel<<<(int)((nW + 255) / 256), 256, 0, stream>>>(Wq, wqb, (int)nW);
  f32_to_bf16_kernel<<<(int)((nW + 255) / 256), 256, 0, stream>>>(Wk, wkb, (int)nW);
  f32_to_bf16_kernel<<<(int)((nW + 255) / 256), 256, 0, stream>>>(Wv, wvb, (int)nW);
  f32_to_bf16_kernel<<<(int)((nW + 255) / 256), 256, 0, stream>>>(Wo, wob, (int)nW);

  // tiles: (12288/32) * (768/32) = 9216 waves -> 2304 blocks of 4 waves
  gemm_bf16_kernel<<<2304, 128, 0, stream>>>(xb, wqb, bq, qb,  NTOK, CC, CC, 0);
  gemm_bf16_kernel<<<2304, 128, 0, stream>>>(xb, wkb, bk, kb,  NTOK, CC, CC, 1);
  gemm_bf16_kernel<<<2304, 128, 0, stream>>>(xb, wvb, bv, vtb, NTOK, CC, CC, 2);

  // 144 (b,h) * 64 q-blocks = 9216 waves -> 2304 blocks
  attn_kernel<<<2304, 128, 0, stream>>>(qb, kb, vtb, yb);

  gemm_bf16_kernel<<<2304, 128, 0, stream>>>(yb, wob, bo, out, NTOK, CC, CC, 3);
}
